// AttentionBlock_91336774517485
// MI455X (gfx1250) — matrix operand order repaired and checked by reading, not compiled, not run on hardware
//
#include <hip/hip_runtime.h>

#define S_LEN 2048
#define D_DIM 1024
#define H_NUM 16
#define DK    64
#define B_NUM 2

typedef __attribute__((ext_vector_type(16))) __bf16 v16bf;
typedef __attribute__((ext_vector_type(8)))  __bf16 v8bf;
typedef __attribute__((ext_vector_type(4)))  __bf16 v4bf;
typedef __attribute__((ext_vector_type(8)))  float  v8f;
typedef __attribute__((ext_vector_type(4)))  unsigned u32x4;
typedef __attribute__((ext_vector_type(8)))  unsigned u32x8;

static __device__ __forceinline__ v8f wmma_bf16(v16bf a, v16bf b, v8f c) {
  // D = A(16x32 bf16) x B(32x16 bf16) + C(16x16 f32)
  return __builtin_amdgcn_wmma_f32_16x16x32_bf16(false, a, false, b, (short)0, c,
                                                 false, false);
}

static __device__ __forceinline__ v16bf pack16(v8bf lo, v8bf hi) {
  v16bf r;
#pragma unroll
  for (int e = 0; e < 8; ++e) { r[e] = lo[e]; r[e + 8] = hi[e]; }
  return r;
}

// Low 32 bits of a generic pointer into LDS == workgroup-relative LDS offset
// (flat aperture).
static __device__ __forceinline__ unsigned lds_off(const void* p) {
  return (unsigned)(unsigned long long)p;
}

// Async DMA: 16 bytes per lane, global -> LDS, tracked by ASYNCcnt.
static __device__ __forceinline__ void async_copy16(unsigned lds, const void* gaddr) {
  asm volatile("global_load_async_to_lds_b128 %0, %1, off"
               :: "v"(lds), "v"(gaddr) : "memory");
}

static __device__ __forceinline__ void wait_async0() {
  asm volatile("s_wait_asynccnt 0" ::: "memory");
}

// Tensor Data Mover: 2D tile, descriptor groups 0/1 in SGPRs (VADDR2/3 null).
static __device__ __forceinline__ void tdm_load_2d(u32x4 g0, u32x8 g1) {
  asm volatile("tensor_load_to_lds %0, %1" :: "s"(g0), "s"(g1) : "memory");
}

static __device__ __forceinline__ void wait_tensor0() {
  __builtin_amdgcn_s_wait_tensorcnt(0);
}

// ---------------------------------------------------------------------------
// Kernel 1: fused Q/K/V projection. grid=(N/128, M/128, 3), block=256.
// Y = X @ W^T + b, emitted as bf16 in head-split layout [B,H,S,DK].
// (fp32->bf16 conversion during staging, so this one keeps the VALU path.)
// ---------------------------------------------------------------------------
__global__ __launch_bounds__(256)
void qkv_proj_kernel(const float* __restrict__ qx, const float* __restrict__ kx,
                     const float* __restrict__ vx,
                     const float* __restrict__ Wq, const float* __restrict__ bq,
                     const float* __restrict__ Wk, const float* __restrict__ bk,
                     const float* __restrict__ Wv, const float* __restrict__ bv,
                     __bf16* __restrict__ qh, __bf16* __restrict__ kh,
                     __bf16* __restrict__ vh)
{
  const int which = blockIdx.z;
  const float* X    = (which == 0) ? qx : (which == 1) ? kx : vx;
  const float* W    = (which == 0) ? Wq : (which == 1) ? Wk : Wv;
  const float* bias = (which == 0) ? bq : (which == 1) ? bk : bv;
  __bf16* Y         = (which == 0) ? qh : (which == 1) ? kh : vh;

  const int m0 = blockIdx.y * 128;   // over B*S = 4096
  const int n0 = blockIdx.x * 128;   // over D   = 1024

  __shared__ __align__(32) __bf16 As[2][128][48];  // [m][k], stride 96B
  __shared__ __align__(32) __bf16 Bs[2][128][48];  // [n][k] = rows of W

  const int t    = threadIdx.x;
  const int lane = t & 31;
  const int wave = t >> 5;     // 8 waves
  const int wr   = wave >> 1;  // 0..3: 32-row strip
  const int wc   = wave & 1;   // 0..1: 64-col strip
  const int ln15 = lane & 15;
  const int koff = (lane >> 4) << 3;   // A-frag K select per ISA layout
  const int kb16 = (lane >> 4) << 4;   // unused; B-frag also uses koff

  v8f acc[2][4] = {};

  const int sr = t >> 3;         // staging row 0..31
  const int sc = (t & 7) << 2;   // staging col (floats)

  auto stage = [&](int ks, int buf) {
    const int kbase = ks * 32;
#pragma unroll
    for (int i = 0; i < 4; ++i) {
      const int r = sr + i * 32;
      float4 fa = *(const float4*)(X + (size_t)(m0 + r) * D_DIM + kbase + sc);
      float4 fb = *(const float4*)(W + (size_t)(n0 + r) * D_DIM + kbase + sc);
      v4bf ha, hb;
      ha[0] = (__bf16)fa.x; ha[1] = (__bf16)fa.y; ha[2] = (__bf16)fa.z; ha[3] = (__bf16)fa.w;
      hb[0] = (__bf16)fb.x; hb[1] = (__bf16)fb.y; hb[2] = (__bf16)fb.z; hb[3] = (__bf16)fb.w;
      *(v4bf*)&As[buf][r][sc] = ha;
      *(v4bf*)&Bs[buf][r][sc] = hb;
    }
  };

  stage(0, 0);
  __syncthreads();

  int buf = 0;
  for (int ks = 0; ks < D_DIM / 32; ++ks) {
    if (ks + 1 < D_DIM / 32) stage(ks + 1, buf ^ 1);  // double-buffered prefetch
    v16bf afrag[2], bfrag[4];
#pragma unroll
    for (int i = 0; i < 2; ++i) {
      const __bf16* p = &As[buf][wr * 32 + i * 16 + ln15][0];
      afrag[i] = pack16(*(const v8bf*)(p + koff), *(const v8bf*)(p + 16 + koff));
    }
#pragma unroll
    for (int j = 0; j < 4; ++j)
      bfrag[j] = pack16(*(const v8bf*)&Bs[buf][wc * 64 + j * 16 + ln15][koff], *(const v8bf*)&Bs[buf][wc * 64 + j * 16 + ln15][16 + koff]);
#pragma unroll
    for (int i = 0; i < 2; ++i)
#pragma unroll
      for (int j = 0; j < 4; ++j)
        acc[i][j] = wmma_bf16(afrag[i], bfrag[j], acc[i][j]);
    __syncthreads();
    buf ^= 1;
  }

  // Store with bias, head-split [B,H,S,DK] bf16.
#pragma unroll
  for (int i = 0; i < 2; ++i) {
#pragma unroll
    for (int j = 0; j < 4; ++j) {
      const int e   = n0 + wc * 64 + j * 16 + ln15;
      const float bv_ = bias[e];
      const int hh = e >> 6, dd = e & 63;
#pragma unroll
      for (int r = 0; r < 8; ++r) {
        const int m  = m0 + wr * 32 + i * 16 + r + ((lane >> 4) << 3);
        const int bb = m >> 11, ss = m & (S_LEN - 1);
        Y[((((size_t)bb * H_NUM + hh) * S_LEN) + ss) * DK + dd] =
            (__bf16)(acc[i][j][r] + bv_);
      }
    }
  }
}

// ---------------------------------------------------------------------------
// Kernel 2: fused masked attention for one (b,h) x 128 q-rows.
// scores = scale * Q K^T, p = scores * A^T (A^T[q,k] = A[k][q]), x = p @ V.
// K tile staged by the Tensor Data Mover (TENSORcnt); Q tile via async
// global->LDS DMA (ASYNCcnt). grid=(S/128, B*H), block=256.
// ---------------------------------------------------------------------------
__global__ __launch_bounds__(256)
void attn_kernel(const __bf16* __restrict__ qh, const __bf16* __restrict__ kh,
                 const __bf16* __restrict__ vh, const float* __restrict__ A,
                 __bf16* __restrict__ xbuf)
{
  const int q0 = blockIdx.x * 128;
  const int bh = blockIdx.y;
  const int b = bh >> 4, h = bh & 15;

  __shared__ __align__(32) __bf16 Qs[128][80];    // [q][dk]
  __shared__ __align__(32) __bf16 Ks[128][80];    // [key][dk] (TDM, padded rows)
  __shared__ __align__(32) __bf16 Vs[64][144];    // [dk][key]   (V transposed)
  __shared__ __align__(32) __bf16 Ps[128][136];   // masked scores, bf16

  const size_t base = (size_t)bh * S_LEN * DK;
  const __bf16* Qg = qh + base + (size_t)q0 * DK;
  const __bf16* Kg = kh + base;
  const __bf16* Vg = vh + base;

  const int t    = threadIdx.x;
  const int lane = t & 31;
  const int wave = t >> 5;
  const int wr   = wave >> 1;
  const int wc   = wave & 1;
  const int ln15 = lane & 15;
  const int koff = (lane >> 4) << 3;
  const int kb16 = (lane >> 4) << 4;

  const int r0 = t >> 3;
  const int c0 = (t & 7) << 3;   // bf16 elems, 16B chunks

  // Q tile staged once via async DMA (reused over entire key loop).
#pragma unroll
  for (int i = 0; i < 4; ++i) {
    const int r = r0 + i * 32;
    async_copy16(lds_off(&Qs[r][c0]), Qg + (size_t)r * DK + c0);
  }

  const unsigned ks_lds = lds_off(&Ks[0][0]);

  v8f xacc[4] = {};              // persistent x[16q x 64dk] per wave
  const float scale = 0.125f;    // 1/sqrt(64)

  for (int kb = 0; kb < S_LEN; kb += 128) {
    __syncthreads();             // protect Ks/Vs/Ps from previous iteration

    if (wave == 0) {
      // TDM descriptor: 2D tile [128 keys x 64 dk] bf16 from kh -> Ks.
      // LDS rows padded: 32 DWORDs data + 8 DWORDs pad = 160B pitch.
      const unsigned long long ga =
          (unsigned long long)(const void*)(Kg + (size_t)kb * DK);
      u32x4 g0;
      g0[0] = 1u;                                   // count=1, user mode
      g0[1] = ks_lds;                               // lds_addr
      g0[2] = (unsigned)ga;                         // global_addr[31:0]
      g0[3] = ((unsigned)(ga >> 32) & 0x01FFFFFFu)  // global_addr[56:32]
              | (2u << 30);                         // type=2 ("image")
      u32x8 g1;
      g1[0] = (1u << 16)      // data_size = 2 bytes
            | (1u << 20)      // pad_enable
            | (4u << 22)      // pad_interval: every 32 DWORDs (one 128B row)
            | (7u << 25);     // pad_amount: 8 DWORDs (32B) -> 160B row pitch
      g1[1] = (unsigned)DK << 16;       // tensor_dim0 = 64  (bits 79:48)
      g1[2] = (unsigned)S_LEN << 16;    // tensor_dim1 = 2048 (bits 111:80)
      g1[3] = (unsigned)DK << 16;       // tile_dim0 = 64    (bits 127:112)
      g1[4] = 128u;                     // tile_dim1 = 128   (bits 143:128)
      g1[5] = (unsigned)DK;             // tensor_dim0_stride = 64 (bits 207:160)
      g1[6] = 0u;
      g1[7] = 0u;
      tdm_load_2d(g0, g1);
    }

    {                            // stage V transposed + prefetch mask rows
#pragma unroll
      for (int i = 0; i < 4; ++i) {
        const int r = r0 + i * 32;
        v8bf dv = *(const v8bf*)(Vg + (size_t)(kb + r) * DK + c0);
#pragma unroll
        for (int e = 0; e < 8; ++e) Vs[c0 + e][r] = dv[e];
        __builtin_prefetch(A + (size_t)(kb + r) * S_LEN + q0 + ((t & 7) << 4), 0, 3);
      }
    }
    wait_async0();               // Q tile DMA complete (first iteration)
    wait_tensor0();              // K tile TDM complete (wave 0; others trivial)
    __syncthreads();

    // GEMM1: s[32q x 64key] per wave, K = DK = 64 (2 wmma steps)
    v8f sacc[2][4] = {};
#pragma unroll
    for (int ks = 0; ks < 2; ++ks) {
      v16bf afrag[2], bfrag[4];
#pragma unroll
      for (int i = 0; i < 2; ++i) {
        const __bf16* p = &Qs[wr * 32 + i * 16 + ln15][ks * 32];
        afrag[i] = pack16(*(const v8bf*)(p + koff), *(const v8bf*)(p + 16 + koff));
      }
#pragma unroll
      for (int j = 0; j < 4; ++j)
        bfrag[j] = pack16(*(const v8bf*)&Ks[wc * 64 + j * 16 + ln15][ks * 32 + koff], *(const v8bf*)&Ks[wc * 64 + j * 16 + ln15][ks * 32 + 16 + koff]);
#pragma unroll
      for (int i = 0; i < 2; ++i)
#pragma unroll
        for (int j = 0; j < 4; ++j)
          sacc[i][j] = wmma_bf16(afrag[i], bfrag[j], sacc[i][j]);
    }

    // scale * mask (A^T), round to bf16 into Ps (A-operand layout for GEMM2)
#pragma unroll
    for (int i = 0; i < 2; ++i) {
#pragma unroll
      for (int j = 0; j < 4; ++j) {
        const int keyl = wc * 64 + j * 16 + ln15;
        const float* Acol = A + (size_t)(kb + keyl) * S_LEN + q0;  // A[key][q]
#pragma unroll
        for (int r = 0; r < 8; ++r) {
          const int ql = wr * 32 + i * 16 + r + ((lane >> 4) << 3);
          Ps[ql][keyl] = (__bf16)(sacc[i][j][r] * scale * Acol[ql]);
        }
      }
    }
    __syncthreads();

    // GEMM2: x[16q x 64dk] += Ps[16q x 128key] @ V^T, K = 128 (4 wmma steps)
    {
      const int arow = wave * 16 + ln15;
#pragma unroll
      for (int ks = 0; ks < 4; ++ks) {
        const __bf16* pp = &Ps[arow][ks * 32];
        v16bf pa = pack16(*(const v8bf*)(pp + koff), *(const v8bf*)(pp + 16 + koff));
#pragma unroll
        for (int j = 0; j < 4; ++j) {
          v16bf vb = pack16(*(const v8bf*)&Vs[j * 16 + ln15][ks * 32 + koff], *(const v8bf*)&Vs[j * 16 + ln15][ks * 32 + 16 + koff]);
          xacc[j] = wmma_bf16(pa, vb, xacc[j]);
        }
      }
    }
  }

  // Store x merged back to [B,S,D] bf16 (heads re-interleaved).
  __bf16* Xg = xbuf + ((size_t)b * S_LEN + q0) * D_DIM + h * DK;
#pragma unroll
  for (int j = 0; j < 4; ++j)
#pragma unroll
    for (int r = 0; r < 8; ++r)
      Xg[(size_t)(wave * 16 + r + ((lane >> 4) << 3)) * D_DIM + j * 16 + ln15] =
          (__bf16)xacc[j][r];
}

// ---------------------------------------------------------------------------
// Kernel 3: output projection, fp32 result + bias. grid=(8, 32), block=256.
// bf16 activation tile staged via async global->LDS DMA; weight tile converted.
// ---------------------------------------------------------------------------
__global__ __launch_bounds__(256)
void out_proj_kernel(const __bf16* __restrict__ Xb, const float* __restrict__ Wo,
                     const float* __restrict__ bo, float* __restrict__ out)
{
  const int m0 = blockIdx.y * 128;
  const int n0 = blockIdx.x * 128;

  __shared__ __align__(32) __bf16 As[2][128][48];
  __shared__ __align__(32) __bf16 Bs[2][128][48];

  const int t    = threadIdx.x;
  const int lane = t & 31;
  const int wave = t >> 5;
  const int wr   = wave >> 1;
  const int wc   = wave & 1;
  const int ln15 = lane & 15;
  const int koff = (lane >> 4) << 3;
  const int kb16 = (lane >> 4) << 4;

  v8f acc[2][4] = {};

  const int sr = t >> 3;         // B staging row 0..31
  const int sc = (t & 7) << 2;   // B staging col (floats)
  const int ar = t >> 2;         // A staging row 0..63
  const int ac = (t & 3) << 3;   // A staging col (bf16, 16B chunks)

  auto stage = [&](int ks, int buf) {
    const int kbase = ks * 32;
#pragma unroll
    for (int i = 0; i < 2; ++i) {   // A operand: async DMA, bf16 verbatim
      const int r = ar + i * 64;
      async_copy16(lds_off(&As[buf][r][ac]),
                   Xb + (size_t)(m0 + r) * D_DIM + kbase + ac);
    }
#pragma unroll
    for (int i = 0; i < 4; ++i) {   // B operand: fp32 -> bf16 convert
      const int r = sr + i * 32;
      float4 fb = *(const float4*)(Wo + (size_t)(n0 + r) * D_DIM + kbase + sc);
      v4bf hb;
      hb[0] = (__bf16)fb.x; hb[1] = (__bf16)fb.y; hb[2] = (__bf16)fb.z; hb[3] = (__bf16)fb.w;
      *(v4bf*)&Bs[buf][r][sc] = hb;
    }
  };

  stage(0, 0);
  wait_async0();
  __syncthreads();

  int buf = 0;
  for (int ks = 0; ks < D_DIM / 32; ++ks) {
    if (ks + 1 < D_DIM / 32) stage(ks + 1, buf ^ 1);  // overlaps with WMMA below
    v16bf afrag[2], bfrag[4];
#pragma unroll
    for (int i = 0; i < 2; ++i) {
      const __bf16* p = &As[buf][wr * 32 + i * 16 + ln15][0];
      afrag[i] = pack16(*(const v8bf*)(p + koff), *(const v8bf*)(p + 16 + koff));
    }
#pragma unroll
    for (int j = 0; j < 4; ++j)
      bfrag[j] = pack16(*(const v8bf*)&Bs[buf][wc * 64 + j * 16 + ln15][koff], *(const v8bf*)&Bs[buf][wc * 64 + j * 16 + ln15][16 + koff]);
#pragma unroll
    for (int i = 0; i < 2; ++i)
#pragma unroll
      for (int j = 0; j < 4; ++j)
        acc[i][j] = wmma_bf16(afrag[i], bfrag[j], acc[i][j]);
    wait_async0();
    __syncthreads();
    buf ^= 1;
  }

#pragma unroll
  for (int i = 0; i < 2; ++i) {
#pragma unroll
    for (int j = 0; j < 4; ++j) {
      const int e = n0 + wc * 64 + j * 16 + ln15;
      const float bv_ = bo[e];
#pragma unroll
      for (int r = 0; r < 8; ++r) {
        const int m = m0 + wr * 32 + i * 16 + r + ((lane >> 4) << 3);
        out[(size_t)m * D_DIM + e] = acc[i][j][r] + bv_;
      }
    }
  }
}

// ---------------------------------------------------------------------------
extern "C" void kernel_launch(void* const* d_in, const int* in_sizes, int n_in,
                              void* d_out, int out_size, void* d_ws, size_t ws_size,
                              hipStream_t stream) {
  (void)in_sizes; (void)n_in; (void)out_size; (void)ws_size;
  const float* q  = (const float*)d_in[0];
  const float* k  = (const float*)d_in[1];
  const float* v  = (const float*)d_in[2];
  const float* A  = (const float*)d_in[3];
  const float* Wq = (const float*)d_in[4];
  const float* bq = (const float*)d_in[5];
  const float* Wk = (const float*)d_in[6];
  const float* bk = (const float*)d_in[7];
  const float* Wv = (const float*)d_in[8];
  const float* bv = (const float*)d_in[9];
  const float* Wo = (const float*)d_in[10];
  const float* bo = (const float*)d_in[11];
  float* out = (float*)d_out;

  const size_t HE = (size_t)B_NUM * H_NUM * S_LEN * DK;  // 4M elems
  __bf16* qh = (__bf16*)d_ws;      // 8 MiB each, 32 MiB total
  __bf16* kh = qh + HE;
  __bf16* vh = kh + HE;
  __bf16* xb = vh + HE;

  qkv_proj_kernel<<<dim3(8, 32, 3), 256, 0, stream>>>(
      q, k, v, Wq, bq, Wk, bk, Wv, bv, qh, kh, vh);
  attn_kernel<<<dim3(S_LEN / 128, B_NUM * H_NUM), 256, 0, stream>>>(
      qh, kh, vh, A, xb);
  out_proj_kernel<<<dim3(8, 32), 256, 0, stream>>>(xb, Wo, bo, out);
}
